// NeuroModule_19653770346814
// MI455X (gfx1250) — matrix operand order repaired and checked by reading, not compiled, not run on hardware
//
#include <hip/hip_runtime.h>

// MI455X (gfx1250) implementation.
// Pipeline:
//  k0: repack 4 weight tensors OIHW f32 -> [tap][co][ci] f16 (WMMA B-friendly)
//  kz: zero LIF states v_a, z_a, v_b, z_b
//  k1: single pass over u (120 MB, the HBM roofline term): compute u_bar and
//      emit f16 channels-innermost transpose uT[t][b][h][w][c]; LDS->global via
//      GLOBAL_STORE_ASYNC_FROM_LDS_B128
//  k2/k3 (x30): implicit-GEMM 3x3 convs on v_wmma_f32_16x16x32_f16 with fused
//      leaky-integrate-and-fire epilogues; global->LDS staging uses
//      GLOBAL_LOAD_ASYNC_TO_LDS_B128 + s_wait_asynccnt. k2 uses 2x2 register
//      blocking (4 WMMA per 2 A + 2 B LDS fragment loads).
//  k4: firing-rate means -> f16
//  k5/k6: final rate-equivalent refinement (dual GEMM + clamp) -> d_out

#define T_STEPS 30
#define BATCH   16
#define CH      64
#define HH      32
#define WW      32

typedef __attribute__((ext_vector_type(16))) _Float16 v16h;
typedef __attribute__((ext_vector_type(8)))  _Float16 v8h;
typedef __attribute__((ext_vector_type(8)))  float    v8f;

__device__ __forceinline__ void cp16(_Float16* d, const _Float16* s) {
  *reinterpret_cast<uint4*>(d) = *reinterpret_cast<const uint4*>(s);
}
__device__ __forceinline__ void zr16(_Float16* d) {
  uint4 z; z.x = z.y = z.z = z.w = 0u;
  *reinterpret_cast<uint4*>(d) = z;
}

// ---- CDNA5 async global<->LDS (ASYNCcnt-tracked, no VGPR data movement) ----
// LDS aperture: flat address low 32 bits are the LDS byte offset.
__device__ __forceinline__ unsigned lds_off(const void* p) {
  return (unsigned)(unsigned long long)p;
}
__device__ __forceinline__ void a_g2l128(void* lds, const void* g) {
  asm volatile("global_load_async_to_lds_b128 %0, %1, off"
               :: "v"(lds_off(lds)), "v"(g) : "memory");
}
__device__ __forceinline__ void a_l2g128(void* g, const void* lds) {
  asm volatile("global_store_async_from_lds_b128 %0, %1, off"
               :: "v"(g), "v"(lds_off(lds)) : "memory");
}
__device__ __forceinline__ void wait_async0() {
  asm volatile("s_wait_asynccnt 0" ::: "memory");
}

// A-fragment (16x32 f16): lane<16 holds M=lane, K=0..15;
// lane>=16 holds M=lane-16, K=16..31. p points at the pixel's ci0
// channel group (contiguous channels in LDS), 16B-aligned.
__device__ __forceinline__ v16h a_frag(const _Float16* p, int hi) {
  const v8h* q = reinterpret_cast<const v8h*>(p);
  v8h lo = q[2 * hi];
  v8h hh = q[2 * hi + 1];
  return __builtin_shufflevector(lo, hh, 0,1,2,3,4,5,6,7,8,9,10,11,12,13,14,15);
}
// B-fragment (32x16 f16): lane n holds column N=n&15, K = 16*(n>=16) .. +15.
// Weights packed [tap][co][ci] so this is one contiguous 32B LDS load.
__device__ __forceinline__ v16h b_frag(const _Float16* wl, int tap, int co, int ci0, int hi) {
  return *reinterpret_cast<const v16h*>(wl + (((tap << 6) + co) << 6) + ci0 + (hi << 4));
}
__device__ __forceinline__ v8f mma(v16h a, v16h b, v8f c) {
  return __builtin_amdgcn_wmma_f32_16x16x32_f16(false, a, false, b, (short)0, c, false, false);
}
// LIF update: v = 0.5v + y; s = (v >= 1); v -= s; z += s
__device__ __forceinline__ void lif(float* __restrict__ v, float* __restrict__ z,
                                    _Float16* __restrict__ s, size_t i, float y) {
  float vv = 0.5f * v[i] + y;
  float sp = vv >= 1.0f ? 1.0f : 0.0f;
  v[i] = vv - sp; z[i] += sp; s[i] = (_Float16)sp;
}
__device__ __forceinline__ void lif_ns(float* __restrict__ v, float* __restrict__ z,
                                       size_t i, float y) {
  float vv = 0.5f * v[i] + y;
  float sp = vv >= 1.0f ? 1.0f : 0.0f;
  v[i] = vv - sp; z[i] += sp;
}

// ---------------- k0: weight repack OIHW f32 -> [tap][co][ci] f16 ----------------
__global__ __launch_bounds__(256) void k0_pack_w(const float* __restrict__ w0,
                                                 const float* __restrict__ w1,
                                                 const float* __restrict__ w2,
                                                 const float* __restrict__ w3,
                                                 _Float16* __restrict__ wp) {
  int idx = blockIdx.x * 256 + threadIdx.x;
  if (idx >= 4 * 36864) return;
  int tsel = idx / 36864;
  int e    = idx % 36864;
  int tap  = e >> 12;        // [0,9)
  int co   = (e >> 6) & 63;
  int ci   = e & 63;
  const float* w = (tsel == 0) ? w0 : (tsel == 1) ? w1 : (tsel == 2) ? w2 : w3;
  wp[idx] = (_Float16)w[(co * 64 + ci) * 9 + tap];
}

// ---------------- kz: zero states ----------------
__global__ __launch_bounds__(256) void k_zero(float4* __restrict__ p, int n4) {
  int i = blockIdx.x * 256 + threadIdx.x;
  if (i < n4) { float4 z; z.x = z.y = z.z = z.w = 0.f; p[i] = z; }
}

// ---------------- k1: one pass over u: u_bar + f16 transpose ----------------
// grid 512 = B*H, 256 threads, LDS = (30*2048 + 2048) halfs
__global__ __launch_bounds__(256) void k1_time(const float* __restrict__ u,
                                               _Float16* __restrict__ uT,
                                               _Float16* __restrict__ ubar) {
  extern __shared__ _Float16 sm[];
  _Float16* sUT = sm;               // [t][w*64+c]
  _Float16* sUB = sm + 30 * 2048;   // [w*64+c]
  const int tid = threadIdx.x;
  const int b = blockIdx.x >> 5;
  const int h = blockIdx.x & 31;

  float p = 1.f;
  #pragma unroll
  for (int i = 0; i < T_STEPS + 1; ++i) p *= 0.9f;
  const float inv_norm = 0.1f / (1.f - p);   // 1 / ((1-L^(T+1))/(1-L))

  for (int it = 0; it < 8; ++it) {
    const int pos = tid + (it << 8);      // 2048 (c,w) positions per (b,h)
    const int c = pos >> 5;
    const int w = pos & 31;
    const size_t base = ((((size_t)b * CH + c) * HH + h) * WW + w) * T_STEPS;
    float vals[T_STEPS];
    const float2* p2 = reinterpret_cast<const float2*>(u + base);
    #pragma unroll
    for (int k = 0; k < 15; ++k) { float2 v = p2[k]; vals[2*k] = v.x; vals[2*k+1] = v.y; }
    // u_bar = u[T-1] + sum_i L^(i+1) * u[(T-2-i) mod T], normalized
    float acc = vals[T_STEPS - 1];
    float lw = 1.f;
    #pragma unroll
    for (int i = 0; i < T_STEPS; ++i) {
      lw *= 0.9f;
      int j = 28 - i; if (j < 0) j += 30;
      acc += lw * vals[j];
    }
    sUB[(w << 6) + c] = (_Float16)(acc * inv_norm);
    #pragma unroll
    for (int t = 0; t < T_STEPS; ++t) sUT[t * 2048 + (w << 6) + c] = (_Float16)vals[t];
  }
  __syncthreads();
  // LDS -> global, asynchronously (GLOBAL_STORE_ASYNC_FROM_LDS_B128)
  for (int t = 0; t < T_STEPS; ++t)
    a_l2g128(uT + (((size_t)t * BATCH + b) * HH + h) * 2048 + (tid << 3),
             sUT + t * 2048 + (tid << 3));
  a_l2g128(ubar + ((size_t)b * HH + h) * 2048 + (tid << 3), sUB + (tid << 3));
  wait_async0();
}

// ---------------- k2: stride-1 conv(u_t, W_in) + LIF on v_a ----------------
// grid 128 = B * (H/4); 8 waves; M=128 px (4 rows) x N=64 ch; 2x2 tiles/wave
__global__ __launch_bounds__(256) void k2_conv_a(const _Float16* __restrict__ uT,
                                                 const _Float16* __restrict__ wp,
                                                 float* __restrict__ v_a,
                                                 float* __restrict__ z_a,
                                                 _Float16* __restrict__ s_a,
                                                 int t) {
  extern __shared__ _Float16 sm[];
  _Float16* wl = sm;            // 36864 halfs
  _Float16* in = sm + 36864;    // 6 rows * 34 w * 64 c = 13056 halfs
  const int tid = threadIdx.x;
  const int b  = blockIdx.x >> 3;
  const int h0 = (blockIdx.x & 7) << 2;

  // weights -> LDS (async, 4608 x 16B chunks)
  #pragma unroll
  for (int i = 0; i < 18; ++i) { int idx = ((i << 8) + tid) << 3; a_g2l128(wl + idx, wp + idx); }

  // input halo rows (h0-1 .. h0+4), width pad 1 both sides (async per row)
  const int dw = tid >> 3, dc = (tid & 7) << 3;
  #pragma unroll
  for (int r = 0; r < 6; ++r) {
    const int ih = h0 - 1 + r;
    _Float16* d = in + ((r * 34 + 1 + dw) << 6) + dc;
    if (ih >= 0 && ih < HH)
      a_g2l128(d, uT + (((size_t)t * BATCH + b) * HH + ih) * 2048 + (tid << 3));
    else
      zr16(d);
  }
  if (tid < 96) {                                  // zero pad cols 0 and 33
    const int r = tid >> 4;
    const int wi = ((tid >> 3) & 1) ? 33 : 0;
    zr16(in + ((r * 34 + wi) << 6) + ((tid & 7) << 3));
  }
  if (t + 1 < T_STEPS)   // warm GL2 for next step's slab (global_prefetch_b8)
    __builtin_prefetch(uT + (((size_t)(t + 1) * BATCH + b) * HH + h0) * 2048 + (tid << 3), 0, 0);
  wait_async0();
  __syncthreads();

  const int wv = tid >> 5, lane = tid & 31, ml = lane & 15, hi = lane >> 4;
  const int mt0 = (wv & 3) << 1, nt0 = (wv >> 2) << 1;   // 2 M-tiles x 2 N-tiles
  const int p0 = (mt0 << 4) + ml, p1 = p0 + 16;
  const int dh0 = p0 >> 5, pw0 = p0 & 31;
  const int dh1 = p1 >> 5, pw1 = p1 & 31;
  v8f acc00 = {}, acc01 = {}, acc10 = {}, acc11 = {};
  #pragma unroll
  for (int tap = 0; tap < 9; ++tap) {
    const int kh = tap / 3, kw = tap % 3;
    const _Float16* ab0 = in + (((dh0 + kh) * 34 + (pw0 + kw)) << 6);
    const _Float16* ab1 = in + (((dh1 + kh) * 34 + (pw1 + kw)) << 6);
    #pragma unroll
    for (int ci0 = 0; ci0 < 64; ci0 += 32) {
      v16h af0 = a_frag(ab0 + ci0, hi);
      v16h af1 = a_frag(ab1 + ci0, hi);
      v16h bf0 = b_frag(wl, tap, (nt0 << 4) + ml, ci0, hi);
      v16h bf1 = b_frag(wl, tap, ((nt0 + 1) << 4) + ml, ci0, hi);
      acc00 = mma(af0, bf0, acc00);
      acc01 = mma(af0, bf1, acc01);
      acc10 = mma(af1, bf0, acc10);
      acc11 = mma(af1, bf1, acc11);
    }
  }
  // LIF epilogue: v=0.5v+y; s=(v>=1); v-=s; z+=s; emit s (f16, [b][h][w][c])
  const int co0 = (nt0 << 4) + ml, co1 = co0 + 16;
  #pragma unroll
  for (int j = 0; j < 8; ++j) {
    {
      const int pp = (mt0 << 4) + j + (hi << 3);
      const int h = h0 + (pp >> 5), w = pp & 31;
      const size_t ib = ((((size_t)b * HH + h) * WW + w) << 6);
      lif(v_a, z_a, s_a, ib + co0, acc00[j]);
      lif(v_a, z_a, s_a, ib + co1, acc01[j]);
    }
    {
      const int pp = ((mt0 + 1) << 4) + j + (hi << 3);
      const int h = h0 + (pp >> 5), w = pp & 31;
      const size_t ib = ((((size_t)b * HH + h) * WW + w) << 6);
      lif(v_a, z_a, s_a, ib + co0, acc10[j]);
      lif(v_a, z_a, s_a, ib + co1, acc11[j]);
    }
  }
}

// ---------------- k3: stride-2 conv(s_a, W_down) + LIF on v_b ----------------
// grid 64 = B * (16/4); M=64 px (4 output rows), N=64 ch. SAME pad: 0 begin, 1 end.
__global__ __launch_bounds__(256) void k3_conv_b(const _Float16* __restrict__ s_a,
                                                 const _Float16* __restrict__ wp,
                                                 float* __restrict__ v_b,
                                                 float* __restrict__ z_b) {
  extern __shared__ _Float16 sm[];
  _Float16* wl = sm;
  _Float16* in = sm + 36864;    // 9 rows * 34 * 64 = 19584 halfs
  const int tid = threadIdx.x;
  const int b   = blockIdx.x >> 2;
  const int ho0 = (blockIdx.x & 3) << 2;
  #pragma unroll
  for (int i = 0; i < 18; ++i) { int idx = ((i << 8) + tid) << 3; a_g2l128(wl + idx, wp + idx); }
  const int dw = tid >> 3, dc = (tid & 7) << 3;
  #pragma unroll
  for (int r = 0; r < 9; ++r) {                    // rows 2*ho0 .. 2*ho0+8
    const int ih = (ho0 << 1) + r;
    _Float16* d = in + ((r * 34 + dw) << 6) + dc;
    if (ih < HH) a_g2l128(d, s_a + (((size_t)b * HH + ih) << 11) + (tid << 3));
    else         zr16(d);
  }
  if (tid < 144) {                                 // zero pad cols 32,33
    const int r = tid >> 4;
    const int wi = 32 + ((tid >> 3) & 1);
    zr16(in + ((r * 34 + wi) << 6) + ((tid & 7) << 3));
  }
  wait_async0();
  __syncthreads();

  const int wv = tid >> 5, lane = tid & 31, ml = lane & 15, hi = lane >> 4;
  const int mt = wv & 3, nt0 = (wv >> 2) << 1;
  v8f acc0 = {}, acc1 = {};
  #pragma unroll
  for (int tap = 0; tap < 9; ++tap) {
    const int kh = tap / 3, kw = tap % 3;
    const _Float16* ab = in + ((((mt << 1) + kh) * 34 + (ml << 1) + kw) << 6);
    #pragma unroll
    for (int ci0 = 0; ci0 < 64; ci0 += 32) {
      v16h af = a_frag(ab + ci0, hi);
      v16h b0 = b_frag(wl, tap, (nt0 << 4) + ml, ci0, hi);
      v16h b1 = b_frag(wl, tap, ((nt0 + 1) << 4) + ml, ci0, hi);
      acc0 = mma(af, b0, acc0);
      acc1 = mma(af, b1, acc1);
    }
  }
  const int ho = ho0 + mt;
  #pragma unroll
  for (int j = 0; j < 8; ++j) {
    const int wo = j + (hi << 3);
    const size_t ib = ((((size_t)b * 16 + ho) * 16 + wo) << 6);
    const int co0 = (nt0 << 4) + ml, co1 = co0 + 16;
    lif_ns(v_b, z_b, ib + co0, acc0[j]);
    lif_ns(v_b, z_b, ib + co1, acc1[j]);
  }
}

// ---------------- k4: firing-rate means -> f16 ----------------
__global__ __launch_bounds__(256) void k4_rates(const float* __restrict__ za,
                                                const float* __restrict__ zb,
                                                _Float16* __restrict__ zah,
                                                _Float16* __restrict__ zbh) {
  int i = blockIdx.x * 256 + threadIdx.x;
  const float inv = 1.0f / 30.0f;
  if (i < 1048576) zah[i] = (_Float16)(za[i] * inv);
  else if (i < 1048576 + 262144) { int k = i - 1048576; zbh[k] = (_Float16)(zb[k] * inv); }
}

// ---------------- k5: a = clip(conv(u_bar,W_in) + conv(z_a,W_aa), 0, 1) ----------------
__global__ __launch_bounds__(256) void k5_final_a(const _Float16* __restrict__ ubar,
                                                  const _Float16* __restrict__ zah,
                                                  const _Float16* __restrict__ wpIn,
                                                  const _Float16* __restrict__ wpAa,
                                                  float* __restrict__ outA,
                                                  _Float16* __restrict__ a_h) {
  extern __shared__ _Float16 sm[];
  _Float16* wlI = sm;
  _Float16* wlA = sm + 36864;
  _Float16* inU = sm + 2 * 36864;            // 8704 halfs
  _Float16* inZ = sm + 2 * 36864 + 8704;     // 8704 halfs
  const int tid = threadIdx.x;
  const int b  = blockIdx.x >> 4;
  const int h0 = (blockIdx.x & 15) << 1;
  #pragma unroll
  for (int i = 0; i < 18; ++i) {
    int idx = ((i << 8) + tid) << 3;
    a_g2l128(wlI + idx, wpIn + idx);
    a_g2l128(wlA + idx, wpAa + idx);
  }
  const int dw = tid >> 3, dc = (tid & 7) << 3;
  #pragma unroll
  for (int r = 0; r < 4; ++r) {
    const int ih = h0 - 1 + r;
    _Float16* dU = inU + ((r * 34 + 1 + dw) << 6) + dc;
    _Float16* dZ = inZ + ((r * 34 + 1 + dw) << 6) + dc;
    if (ih >= 0 && ih < HH) {
      const size_t rb = (((size_t)b * HH + ih) << 11) + (tid << 3);
      a_g2l128(dU, ubar + rb);
      a_g2l128(dZ, zah + rb);
    } else { zr16(dU); zr16(dZ); }
  }
  if (tid < 64) {
    const int r = tid >> 4;
    const int wi = ((tid >> 3) & 1) ? 33 : 0;
    zr16(inU + ((r * 34 + wi) << 6) + ((tid & 7) << 3));
    zr16(inZ + ((r * 34 + wi) << 6) + ((tid & 7) << 3));
  }
  wait_async0();
  __syncthreads();

  const int wv = tid >> 5, lane = tid & 31, ml = lane & 15, hi = lane >> 4;
  const int mt = wv & 3, nt0 = (wv >> 2) << 1;
  const int pp = (mt << 4) + ml, dh = pp >> 5, pw = pp & 31;
  v8f acc0 = {}, acc1 = {};
  #pragma unroll
  for (int tap = 0; tap < 9; ++tap) {
    const int kh = tap / 3, kw = tap % 3;
    const int off = ((dh + kh) * 34 + (pw + kw)) << 6;
    #pragma unroll
    for (int ci0 = 0; ci0 < 64; ci0 += 32) {
      v16h aU = a_frag(inU + off + ci0, hi);
      v16h b0 = b_frag(wlI, tap, (nt0 << 4) + ml, ci0, hi);
      v16h b1 = b_frag(wlI, tap, ((nt0 + 1) << 4) + ml, ci0, hi);
      acc0 = mma(aU, b0, acc0);
      acc1 = mma(aU, b1, acc1);
      v16h aZ = a_frag(inZ + off + ci0, hi);
      v16h c0 = b_frag(wlA, tap, (nt0 << 4) + ml, ci0, hi);
      v16h c1 = b_frag(wlA, tap, ((nt0 + 1) << 4) + ml, ci0, hi);
      acc0 = mma(aZ, c0, acc0);
      acc1 = mma(aZ, c1, acc1);
    }
  }
  #pragma unroll
  for (int j = 0; j < 8; ++j) {
    const int pp2 = (mt << 4) + j + (hi << 3);
    const int h = h0 + (pp2 >> 5);
    const int w = pp2 & 31;
    const int co0 = (nt0 << 4) + ml, co1 = co0 + 16;
    float r0 = fminf(fmaxf(acc0[j], 0.f), 1.f);
    float r1 = fminf(fmaxf(acc1[j], 0.f), 1.f);
    outA[(((size_t)b * CH + co0) * HH + h) * WW + w] = r0;   // NCHW f32
    outA[(((size_t)b * CH + co1) * HH + h) * WW + w] = r1;
    const size_t ib = ((((size_t)b * HH + h) * WW + w) << 6);
    a_h[ib + co0] = (_Float16)r0;                            // conv input for k6
    a_h[ib + co1] = (_Float16)r1;
  }
}

// ---------------- k6: b = clip(conv(a,W_down,s2) + conv(z_b,W_bb), 0, 1) ----------------
__global__ __launch_bounds__(256) void k6_final_b(const _Float16* __restrict__ a_h,
                                                  const _Float16* __restrict__ zbh,
                                                  const _Float16* __restrict__ wpDn,
                                                  const _Float16* __restrict__ wpBb,
                                                  float* __restrict__ outB) {
  extern __shared__ _Float16 sm[];
  _Float16* wlD = sm;
  _Float16* wlB = sm + 36864;
  _Float16* inA = sm + 2 * 36864;            // 9*34*64 = 19584 halfs
  _Float16* inZ = sm + 2 * 36864 + 19584;    // 6*18*64 = 6912 halfs
  const int tid = threadIdx.x;
  const int b   = blockIdx.x >> 2;
  const int ho0 = (blockIdx.x & 3) << 2;
  #pragma unroll
  for (int i = 0; i < 18; ++i) {
    int idx = ((i << 8) + tid) << 3;
    a_g2l128(wlD + idx, wpDn + idx);
    a_g2l128(wlB + idx, wpBb + idx);
  }
  const int dw = tid >> 3, dc = (tid & 7) << 3;
  #pragma unroll
  for (int r = 0; r < 9; ++r) {              // a_h rows 2*ho0 .. 2*ho0+8
    const int ih = (ho0 << 1) + r;
    _Float16* d = inA + ((r * 34 + dw) << 6) + dc;
    if (ih < HH) a_g2l128(d, a_h + (((size_t)b * HH + ih) << 11) + (tid << 3));
    else         zr16(d);
  }
  if (tid < 144) {
    const int r = tid >> 4;
    const int wi = 32 + ((tid >> 3) & 1);
    zr16(inA + ((r * 34 + wi) << 6) + ((tid & 7) << 3));
  }
  if (tid < 128) {                           // z_b rows ho0-1 .. ho0+4, left pad 1
    const int zw = tid >> 3, zc = (tid & 7) << 3;
    #pragma unroll
    for (int r = 0; r < 6; ++r) {
      const int ih = ho0 - 1 + r;
      _Float16* d = inZ + ((r * 18 + 1 + zw) << 6) + zc;
      if (ih >= 0 && ih < 16)
        a_g2l128(d, zbh + ((((size_t)b * 16 + ih) * 16 + zw) << 6) + zc);
      else zr16(d);
    }
  }
  if (tid < 96) {                            // zero pad cols 0, 17
    const int r = tid >> 4;
    const int wi = ((tid >> 3) & 1) ? 17 : 0;
    zr16(inZ + ((r * 18 + wi) << 6) + ((tid & 7) << 3));
  }
  wait_async0();
  __syncthreads();

  const int wv = tid >> 5, lane = tid & 31, ml = lane & 15, hi = lane >> 4;
  const int mt = wv & 3, nt0 = (wv >> 2) << 1;
  v8f acc0 = {}, acc1 = {};
  #pragma unroll
  for (int tap = 0; tap < 9; ++tap) {
    const int kh = tap / 3, kw = tap % 3;
    const int offA = (((mt << 1) + kh) * 34 + (ml << 1) + kw) << 6;  // stride-2 part
    const int offZ = ((mt + kh) * 18 + ml + kw) << 6;                // stride-1 part
    #pragma unroll
    for (int ci0 = 0; ci0 < 64; ci0 += 32) {
      v16h aA = a_frag(inA + offA + ci0, hi);
      v16h b0 = b_frag(wlD, tap, (nt0 << 4) + ml, ci0, hi);
      v16h b1 = b_frag(wlD, tap, ((nt0 + 1) << 4) + ml, ci0, hi);
      acc0 = mma(aA, b0, acc0);
      acc1 = mma(aA, b1, acc1);
      v16h aZ = a_frag(inZ + offZ + ci0, hi);
      v16h c0 = b_frag(wlB, tap, (nt0 << 4) + ml, ci0, hi);
      v16h c1 = b_frag(wlB, tap, ((nt0 + 1) << 4) + ml, ci0, hi);
      acc0 = mma(aZ, c0, acc0);
      acc1 = mma(aZ, c1, acc1);
    }
  }
  const int ho = ho0 + mt;
  #pragma unroll
  for (int j = 0; j < 8; ++j) {
    const int wo = j + (hi << 3);
    const int co0 = (nt0 << 4) + ml, co1 = co0 + 16;
    outB[(((size_t)b * CH + co0) * 16 + ho) * 16 + wo] = fminf(fmaxf(acc0[j], 0.f), 1.f);
    outB[(((size_t)b * CH + co1) * 16 + ho) * 16 + wo] = fminf(fmaxf(acc1[j], 0.f), 1.f);
  }
}

extern "C" void kernel_launch(void* const* d_in, const int* in_sizes, int n_in,
                              void* d_out, int out_size, void* d_ws, size_t ws_size,
                              hipStream_t stream) {
  (void)in_sizes; (void)n_in; (void)out_size; (void)ws_size;
  const float* u    = (const float*)d_in[0];
  const float* W_in = (const float*)d_in[1];
  const float* W_aa = (const float*)d_in[2];
  const float* W_dn = (const float*)d_in[3];
  const float* W_bb = (const float*)d_in[4];
  char* ws = (char*)d_ws;

  // Workspace layout (bytes). States first (zeroed as one contiguous 10 MB range).
  float*    v_a  = (float*)(ws + 0);              //  4 MB  [b][h][w][c] f32
  float*    z_a  = (float*)(ws + 4194304);        //  4 MB
  float*    v_b  = (float*)(ws + 8388608);        //  1 MB
  float*    z_b  = (float*)(ws + 9437184);        //  1 MB
  _Float16* uT   = (_Float16*)(ws + 10485760);    // 60 MB  [t][b][h][w][c] f16
  _Float16* ubar = (_Float16*)(ws + 73400320);    //  2 MB
  _Float16* s_a  = (_Float16*)(ws + 75497472);    //  2 MB
  _Float16* zah  = (_Float16*)(ws + 77594624);    //  2 MB
  _Float16* zbh  = (_Float16*)(ws + 79691776);    //  0.5 MB
  _Float16* a_h  = (_Float16*)(ws + 80216064);    //  2 MB
  _Float16* wp   = (_Float16*)(ws + 82313216);    //  288 KB, 4x [tap][co][ci] f16
  _Float16* wpIn = wp;
  _Float16* wpAa = wp + 36864;
  _Float16* wpDn = wp + 2 * 36864;
  _Float16* wpBb = wp + 3 * 36864;

  float* outA = (float*)d_out;          // a: [16,64,32,32] f32
  float* outB = outA + 1048576;         // b: [16,64,16,16] f32

  k0_pack_w<<<576, 256, 0, stream>>>(W_in, W_aa, W_dn, W_bb, wp);
  k_zero<<<2560, 256, 0, stream>>>((float4*)ws, 655360);   // zero v_a,z_a,v_b,z_b

  const int SM1 = (30 * 2048 + 2048) * 2;                  // 126976 B
  k1_time<<<512, 256, SM1, stream>>>(u, uT, ubar);

  const int SM2 = (36864 + 13056) * 2;                     //  99840 B
  const int SM3 = (36864 + 19584) * 2;                     // 112896 B
  for (int t = 0; t < 30; ++t) {
    k2_conv_a<<<128, 256, SM2, stream>>>(uT, wpIn, v_a, z_a, s_a, t);
    k3_conv_b<<<64, 256, SM3, stream>>>(s_a, wpDn, v_b, z_b);
  }

  k4_rates<<<5120, 256, 0, stream>>>(z_a, z_b, zah, zbh);

  const int SM4 = (2 * 36864 + 2 * 8704) * 2;              // 182272 B
  k5_final_a<<<256, 256, SM4, stream>>>(ubar, zah, wpIn, wpAa, outA, a_h);

  const int SM5 = (2 * 36864 + 19584 + 6912) * 2;          // 200448 B
  k6_final_b<<<64, 256, SM5, stream>>>(a_h, zbh, wpDn, wpBb, outB);
}
